// NaFlexEmbeds_28896539968317
// MI455X (gfx1250) — matrix operand order repaired and checked by reading, not compiled, not run on hardware
//
#include <hip/hip_runtime.h>
#include <cstdint>

// ---------------------------------------------------------------------------
// NaFlex embed: out = patches @ proj_w + bias + masked bilinear pos-embed
// GEMM: bf16x3 split-precision (hi*hi + lo*hi + hi*lo) on
// v_wmma_f32_16x16x32_bf16 (wave32). Wave tile 16x64, 12 WMMA per k-step.
// Software-pipelined: next k-step's A/B loads are issued before the current
// k-step's WMMAs so no s_wait exposes raw memory latency.
// ---------------------------------------------------------------------------

typedef __attribute__((ext_vector_type(16))) __bf16       v16bf;
typedef __attribute__((ext_vector_type(8)))  float        v8f;
typedef __attribute__((ext_vector_type(8)))  unsigned int v8u;

#define DIN   768
#define CH    1152
#define SEQL  1024
#define NIMG  64
#define KT_N  (DIN / 32)   // 24 k-tiles
#define NT_N  (CH / 16)    // 72 n-tiles
#define TILE_U32 256       // one packed B tile: 32 lanes * 8 dwords
#define BHI_OFF 256        // u32 offset of hi-packed W in ws (after grid buf)
#define BLO_OFF (BHI_OFF + KT_N * NT_N * TILE_U32)

// Fast split of two f32 into packed bf16 hi dword + packed bf16 lo dword.
// hi = truncate-to-bf16, lo = truncate(a - hi). Pairing via one v_perm_b32.
__device__ __forceinline__ void splitp(float a, float b, unsigned &h, unsigned &l) {
  unsigned ua = __builtin_bit_cast(unsigned, a);
  unsigned ub = __builtin_bit_cast(unsigned, b);
  h = __builtin_amdgcn_perm(ub, ua, 0x07060302u);   // {hi16(b), hi16(a)}
  float ra = a - __builtin_bit_cast(float, ua & 0xFFFF0000u);
  float rb = b - __builtin_bit_cast(float, ub & 0xFFFF0000u);
  l = __builtin_amdgcn_perm(__builtin_bit_cast(unsigned, rb),
                            __builtin_bit_cast(unsigned, ra), 0x07060302u);
}

// --------------------------- kernel 1: grid = max(coord)+1 ------------------
__global__ __launch_bounds__(256) void grid_kernel(const int* __restrict__ coord,
                                                   int* __restrict__ gridbuf) {
  __shared__ int smy[256];
  __shared__ int smx[256];
  const int n = blockIdx.x, t = threadIdx.x;
  int my = 0, mx = 0;
  for (int i = t; i < SEQL; i += 256) {
    my = max(my, coord[(n * SEQL + i) * 2 + 0]);
    mx = max(mx, coord[(n * SEQL + i) * 2 + 1]);
  }
  smy[t] = my; smx[t] = mx;
  __syncthreads();
  for (int off = 128; off > 0; off >>= 1) {
    if (t < off) { smy[t] = max(smy[t], smy[t + off]); smx[t] = max(smx[t], smx[t + off]); }
    __syncthreads();
  }
  if (t == 0) { gridbuf[n * 2 + 0] = smy[0] + 1; gridbuf[n * 2 + 1] = smx[0] + 1; }
}

// --------- kernel 2: pack proj_w into WMMA B-operand layout (hi + lo) -------
// B in this code's contiguous k order (32x16, 8 "VGPR" dword slots): lanes 0-15 hold K=0..15
// (2 per slot), lanes 16-31 hold K=16..31; column = lane & 15.
__global__ __launch_bounds__(256) void pack_w(const float* __restrict__ w,
                                              unsigned* __restrict__ ws) {
  const int tid = blockIdx.x * 256 + threadIdx.x;
  if (tid >= KT_N * NT_N * 32) return;
  const int lane = tid & 31;
  const int tile = tid >> 5;               // tileIdx = kt * NT_N + nt
  const int kt = tile / NT_N, nt = tile % NT_N;
  const int n  = nt * 16 + (lane & 15);
  const int kb = kt * 32 + (lane >> 4) * 16;
  unsigned* hi = ws + BHI_OFF + tile * TILE_U32 + lane * 8;
  unsigned* lo = ws + BLO_OFF + tile * TILE_U32 + lane * 8;
#pragma unroll
  for (int v = 0; v < 8; ++v) {
    float a = w[(size_t)(kb + 2 * v + 0) * CH + n];
    float b = w[(size_t)(kb + 2 * v + 1) * CH + n];
    unsigned h, l; splitp(a, b, h, l);
    hi[v] = h; lo[v] = l;
  }
}

// ---------------- kernel 3: fused GEMM (bf16x3 WMMA) + epilogue -------------
// Block = 256 thr = 8 waves stacked in M; wave tile = 16(M) x 64(N).
// Block tile = 128(M) x 64(N); grid = (1152/64, 65536/128) = (18, 512).
__global__ __launch_bounds__(256, 1) void gemm_epilogue(
    const float* __restrict__ patches, const float* __restrict__ pos_embed,
    const float* __restrict__ proj_b,  const unsigned* __restrict__ ws,
    const int* __restrict__ gridbuf,   float* __restrict__ out) {
  const int lane = threadIdx.x & 31;
  const int wave = threadIdx.x >> 5;           // 0..7, one 16-row M tile each
  const int l16 = lane & 15, hl = lane >> 4;

  const int mWave = blockIdx.y * 128 + wave * 16;
  const int nBase = blockIdx.x * 64;
  const int ntile0 = nBase >> 4;               // 4 consecutive n-tiles

  // A fragment in this code's contiguous k order (16x32): row M = lane&15; lanes<16 carry
  // K 0..15, lanes>=16 carry K 16..31.
  const float* aRowP = patches + (size_t)(mWave + l16) * DIN + hl * 16;
  const unsigned* bHiP = ws + BHI_OFF + (size_t)ntile0 * TILE_U32 + lane * 8;
  const unsigned* bLoP = ws + BLO_OFF + (size_t)ntile0 * TILE_U32 + lane * 8;

  v8f acc[4] = {};

  // ---- software pipeline: preload k-step 0 ----
  float4 a0, a1, a2, a3;
  v8u bh[4], bl[4];
  {
    const float4* p = reinterpret_cast<const float4*>(aRowP);
    a0 = p[0]; a1 = p[1];
    const float4* q = reinterpret_cast<const float4*>(aRowP + 8);
    a2 = q[0]; a3 = q[1];
#pragma unroll
    for (int ni = 0; ni < 4; ++ni) {
      bh[ni] = *reinterpret_cast<const v8u*>(bHiP + ni * TILE_U32);
      bl[ni] = *reinterpret_cast<const v8u*>(bLoP + ni * TILE_U32);
    }
  }

#pragma unroll
  for (int kt = 0; kt < KT_N; ++kt) {
    // capture current generation (SSA renames, no copies emitted)
    float4 c0 = a0, c1 = a1, c2 = a2, c3 = a3;
    v8u cbh0 = bh[0], cbh1 = bh[1], cbh2 = bh[2], cbh3 = bh[3];
    v8u cbl0 = bl[0], cbl1 = bl[1], cbl2 = bl[2], cbl3 = bl[3];

    // ---- issue next k-step's loads before this k-step's math ----
    if (kt + 1 < KT_N) {
      const float4* p = reinterpret_cast<const float4*>(aRowP + (kt + 1) * 32);
      a0 = p[0]; a1 = p[1];
      const float4* q = reinterpret_cast<const float4*>(aRowP + (kt + 1) * 32 + 8);
      a2 = q[0]; a3 = q[1];
      const int koff = (kt + 1) * NT_N * TILE_U32;
#pragma unroll
      for (int ni = 0; ni < 4; ++ni) {
        bh[ni] = *reinterpret_cast<const v8u*>(bHiP + koff + ni * TILE_U32);
        bl[ni] = *reinterpret_cast<const v8u*>(bLoP + koff + ni * TILE_U32);
      }
    }

    // ---- split A into bf16 hi/lo fragments ----
    unsigned h0,h1,h2,h3,h4,h5,h6,h7, l0,l1,l2,l3,l4,l5,l6,l7;
    splitp(c0.x, c0.y, h0, l0); splitp(c0.z, c0.w, h1, l1);
    splitp(c1.x, c1.y, h2, l2); splitp(c1.z, c1.w, h3, l3);
    splitp(c2.x, c2.y, h4, l4); splitp(c2.z, c2.w, h5, l5);
    splitp(c3.x, c3.y, h6, l6); splitp(c3.z, c3.w, h7, l7);
    v8u hv = {h0, h1, h2, h3, h4, h5, h6, h7};
    v8u lv = {l0, l1, l2, l3, l4, l5, l6, l7};
    const v16bf ahi = __builtin_bit_cast(v16bf, hv);
    const v16bf alo = __builtin_bit_cast(v16bf, lv);

    const v16bf vbh[4] = {__builtin_bit_cast(v16bf, cbh0), __builtin_bit_cast(v16bf, cbh1),
                          __builtin_bit_cast(v16bf, cbh2), __builtin_bit_cast(v16bf, cbh3)};
    const v16bf vbl[4] = {__builtin_bit_cast(v16bf, cbl0), __builtin_bit_cast(v16bf, cbl1),
                          __builtin_bit_cast(v16bf, cbl2), __builtin_bit_cast(v16bf, cbl3)};

    // ---- bf16x3: hi*hi + lo*hi + hi*lo, fp32 accumulate (12 WMMA) ----
#pragma unroll
    for (int ni = 0; ni < 4; ++ni) {
      acc[ni] = __builtin_amdgcn_wmma_f32_16x16x32_bf16(
          false, ahi, false, vbh[ni], (short)0, acc[ni], false, false);
      acc[ni] = __builtin_amdgcn_wmma_f32_16x16x32_bf16(
          false, alo, false, vbh[ni], (short)0, acc[ni], false, false);
      acc[ni] = __builtin_amdgcn_wmma_f32_16x16x32_bf16(
          false, ahi, false, vbl[ni], (short)0, acc[ni], false, false);
    }
  }

  // ---- epilogue: bias + masked bilinear pos-embed, then store ----
  // C/D layout: slot r of v8f -> M = r + 8*(lane>=16), N = lane&15.
  const int nimg = blockIdx.y >> 3;            // 8 M-blocks per image
  const int gy = gridbuf[nimg * 2 + 0];
  const int gx = gridbuf[nimg * 2 + 1];
  const int seq = gy * gx;
  const float ry = 16.0f / (float)gy;
  const float rx = 16.0f / (float)gx;
  const int sBase = (blockIdx.y & 7) * 128 + wave * 16;

#pragma unroll
  for (int r = 0; r < 8; ++r) {
    const int srow = sBase + hl * 8 + r;       // row within image
    const unsigned rowi = (unsigned)srow / (unsigned)gx;
    const unsigned coli = (unsigned)srow - rowi * (unsigned)gx;
    float sy = fminf(fmaxf(((float)rowi + 0.5f) * ry - 0.5f, 0.0f), 15.0f);
    float sx = fminf(fmaxf(((float)coli + 0.5f) * rx - 0.5f, 0.0f), 15.0f);
    const int y0 = (int)sy, x0 = (int)sx;
    const int y1 = min(y0 + 1, 15), x1 = min(x0 + 1, 15);
    const float wy = sy - (float)y0, wx = sx - (float)x0;
    const float w00 = (1.0f - wy) * (1.0f - wx), w01 = (1.0f - wy) * wx;
    const float w10 = wy * (1.0f - wx),          w11 = wy * wx;
    const bool act = srow < seq;
    const float* p00 = pos_embed + (size_t)(y0 * 16 + x0) * CH;
    const float* p01 = pos_embed + (size_t)(y0 * 16 + x1) * CH;
    const float* p10 = pos_embed + (size_t)(y1 * 16 + x0) * CH;
    const float* p11 = pos_embed + (size_t)(y1 * 16 + x1) * CH;
    const size_t orow = ((size_t)blockIdx.y * 128 + wave * 16 + hl * 8 + r) * CH;
#pragma unroll
    for (int ni = 0; ni < 4; ++ni) {
      const int c = nBase + ni * 16 + l16;
      float pe = act ? (w00 * p00[c] + w01 * p01[c] + w10 * p10[c] + w11 * p11[c])
                     : 0.0f;
      out[orow + c] = acc[ni][r] + proj_b[c] + pe;
    }
  }
}

// ---------------------------------------------------------------------------
extern "C" void kernel_launch(void* const* d_in, const int* in_sizes, int n_in,
                              void* d_out, int out_size, void* d_ws, size_t ws_size,
                              hipStream_t stream) {
  const float* patches = (const float*)d_in[0];
  const int*   coord   = (const int*)d_in[1];
  const float* pose    = (const float*)d_in[2];
  const float* pw      = (const float*)d_in[3];
  const float* pb      = (const float*)d_in[4];
  float*       out     = (float*)d_out;
  unsigned*    ws      = (unsigned*)d_ws;
  int*         gridbuf = (int*)d_ws;     // first 128 u32 of ws

  grid_kernel<<<NIMG, 256, 0, stream>>>(coord, gridbuf);
  pack_w<<<(KT_N * NT_N * 32 + 255) / 256, 256, 0, stream>>>(pw, ws);
  gemm_epilogue<<<dim3(CH / 64, (NIMG * SEQL) / 128), 256, 0, stream>>>(
      patches, pose, pb, ws, gridbuf, out);
}
